// FEMREncoderLayer_1984274891464
// MI455X (gfx1250) — matrix operand order repaired and checked by reading, not compiled, not run on hardware
//
#include <hip/hip_runtime.h>
#include <hip/hip_bf16.h>
#include <math.h>

typedef __attribute__((ext_vector_type(16))) _Float16 v16h;
typedef __attribute__((ext_vector_type(8)))  _Float16 v8h;
typedef __attribute__((ext_vector_type(8)))  float    v8f;

#define B_  2
#define S_  4096
#define H_  1024
#define NH_ 16
#define HD_ 64
#define I_  4096
#define W_  512
#define M_  (B_ * S_)      // 8192 rows
#define QKV_N 3072

__device__ __forceinline__ v16h cat16(v8h lo, v8h hi) {
    return __builtin_shufflevector(lo, hi, 0, 1, 2, 3, 4, 5, 6, 7,
                                   8, 9, 10, 11, 12, 13, 14, 15);
}

// ---------------------------------------------------------------------------
// RMSNorm: x[row,1024] -> xn f16
// ---------------------------------------------------------------------------
__global__ __launch_bounds__(256) void rmsnorm_kernel(const float* __restrict__ x,
                                                      const float* __restrict__ w,
                                                      _Float16* __restrict__ xn) {
    int row = blockIdx.x;
    const float* xr = x + (size_t)row * H_;
    float s = 0.f;
    for (int i = threadIdx.x; i < H_; i += 256) { float v = xr[i]; s += v * v; }
    __shared__ float red[256];
    red[threadIdx.x] = s;
    __syncthreads();
    for (int st = 128; st > 0; st >>= 1) {
        if (threadIdx.x < st) red[threadIdx.x] += red[threadIdx.x + st];
        __syncthreads();
    }
    float inv = rsqrtf(red[0] / (float)H_ + 1e-6f);
    for (int i = threadIdx.x; i < H_; i += 256)
        xn[(size_t)row * H_ + i] = (_Float16)(xr[i] * inv * w[i]);
}

// ---------------------------------------------------------------------------
// fp32 -> f16 elementwise convert
// ---------------------------------------------------------------------------
__global__ __launch_bounds__(256) void cvt_f32_f16(const float* __restrict__ src,
                                                   _Float16* __restrict__ dst, long n) {
    long i = (long)blockIdx.x * 256 + threadIdx.x;
    long stride = (long)gridDim.x * 256;
    for (; i < n; i += stride) dst[i] = (_Float16)src[i];
}

// ---------------------------------------------------------------------------
// V transpose: qkv v-plane [B,S,NH,HD] -> vT [B,NH,HD,S] (d-major for PV frags)
// ---------------------------------------------------------------------------
__global__ __launch_bounds__(256) void transpose_v(const _Float16* __restrict__ qkv,
                                                   _Float16* __restrict__ vT) {
    long idx = (long)blockIdx.x * 256 + threadIdx.x;   // B*NH*HD*S = 8,388,608
    if (idx >= (long)B_ * NH_ * HD_ * S_) return;
    int s = idx & 4095;
    int d = (int)(idx >> 12) & 63;
    int h = (int)(idx >> 18) & 15;
    int b = (int)(idx >> 22);
    vT[idx] = qkv[((size_t)(b * S_ + s)) * QKV_N + 2 * H_ + h * HD_ + d];
}

// ---------------------------------------------------------------------------
// Generic WMMA GEMM: C[M,N] = act(A[M,K] @ B[K,N] + bias), f16 in, f32 acc.
// Block tile 128x128, 8 waves x (32x64), K-step 32.
// A staged via CDNA5 async global->LDS B128; B staged transposed (col-major).
// ---------------------------------------------------------------------------
__device__ __forceinline__ float gelu_exact(float v) {
    return 0.5f * v * (1.0f + erff(v * 0.70710678118654752f));
}

template <int ACT, int ACCUM, bool BIAS, bool OF32, bool OF16>
__global__ __launch_bounds__(256) void gemm_wmma(const _Float16* __restrict__ A,
                                                 const _Float16* __restrict__ Bw,
                                                 const float* __restrict__ bias,
                                                 float* __restrict__ Cf,
                                                 _Float16* __restrict__ Ch,
                                                 int M, int N, int K) {
    __shared__ _Float16 As[128][40];    // row-major: [m][k], 16B-aligned rows
    __shared__ _Float16 Bs[128][40];    // col-major: [n][k], 16B-aligned rows

    const int n0 = blockIdx.x * 128;
    const int m0 = blockIdx.y * 128;
    const int wave = threadIdx.x >> 5;
    const int lane = threadIdx.x & 31;
    const int wm = wave & 3;            // 4 M-groups of 32
    const int wn = wave >> 2;           // 2 N-groups of 64
    const int col = lane & 15;
    const int lh = lane >> 4;

    v8f acc[2][4] = {};

    for (int k0 = 0; k0 < K; k0 += 32) {
        // --- A tile: 512 x 16B async copies, layouts match (K-contiguous) ---
        for (int c = threadIdx.x; c < 512; c += 256) {
            int r = c >> 2, c8 = (c & 3) << 3;
            unsigned lo = (unsigned)(uintptr_t)&As[r][c8];
            const _Float16* src = A + (size_t)(m0 + r) * K + k0 + c8;
            asm volatile("global_load_async_to_lds_b128 %0, %1, off"
                         :: "v"(lo), "v"(src) : "memory");
        }
        // --- B tile: load row-major 16B chunks, scatter to col-major LDS ---
        for (int c = threadIdx.x; c < 512; c += 256) {
            int r = c >> 4, c8 = (c & 15) << 3;      // 32 K-rows x 16 chunks
            v8h bchunk = *(const v8h*)(Bw + (size_t)(k0 + r) * N + n0 + c8);
#pragma unroll
            for (int t = 0; t < 8; ++t) Bs[c8 + t][r] = bchunk[t];
        }
        asm volatile("s_wait_asynccnt 0" ::: "memory");
        __syncthreads();

        v16h af[2];
#pragma unroll
        for (int i = 0; i < 2; ++i) {
            int row = wm * 32 + i * 16 + col;
            int kb = lh ? 8 : 0;
            v8h lo = *(const v8h*)&As[row][kb];
            v8h hi = *(const v8h*)&As[row][kb + 16];
            af[i] = cat16(lo, hi);
        }
        v16h bf[4];
#pragma unroll
        for (int j = 0; j < 4; ++j) {
            int cc = wn * 64 + j * 16 + col;
            int kb = lh ? 8 : 0;
            v8h lo = *(const v8h*)&Bs[cc][kb];
            v8h hi = *(const v8h*)&Bs[cc][kb + 16];
            bf[j] = cat16(lo, hi);
        }
#pragma unroll
        for (int i = 0; i < 2; ++i)
#pragma unroll
            for (int j = 0; j < 4; ++j)
                acc[i][j] = __builtin_amdgcn_wmma_f32_16x16x32_f16(
                    false, af[i], false, bf[j], (short)0, acc[i][j], false, false);
        __syncthreads();
    }

    // --- epilogue (compile-time specialized) ---
    const int rowg = lh ? 8 : 0;
#pragma unroll
    for (int i = 0; i < 2; ++i) {
#pragma unroll
        for (int j = 0; j < 4; ++j) {
            int cc = n0 + wn * 64 + j * 16 + col;
            float bv = BIAS ? bias[cc] : 0.f;
#pragma unroll
            for (int r = 0; r < 8; ++r) {
                int row = m0 + wm * 32 + i * 16 + rowg + r;
                float v = acc[i][j][r] + bv;
                if (ACT == 1) v = gelu_exact(v);
                size_t idx = (size_t)row * N + cc;
                if (ACCUM)      Cf[idx] += v;
                else if (OF32)  Cf[idx] = v;
                if (OF16)       Ch[idx] = (_Float16)v;
            }
        }
    }
}

// ---------------------------------------------------------------------------
// RoPE in-place on f16 qkv buffer [B*S, 3072]: planes q (off 0), k (off 1024)
// ---------------------------------------------------------------------------
__global__ __launch_bounds__(256) void rope_kernel(_Float16* __restrict__ qkv,
                                                   const float* __restrict__ sinb,
                                                   const float* __restrict__ cosb) {
    long idx = (long)blockIdx.x * 256 + threadIdx.x;   // B*S*NH*HD/2 = 4,194,304
    if (idx >= (long)B_ * S_ * NH_ * (HD_ / 2)) return;
    int i = idx & 31;
    int h = (int)(idx >> 5) & 15;
    int s = (int)(idx >> 9) & 4095;
    int b = (int)(idx >> 21);
    size_t rowoff = ((size_t)(b * S_ + s)) * QKV_N + h * HD_ + 2 * i;
    float s0 = sinb[s * HD_ + 2 * i],     c0 = cosb[s * HD_ + 2 * i];
    float s1 = sinb[s * HD_ + 2 * i + 1], c1 = cosb[s * HD_ + 2 * i + 1];
#pragma unroll
    for (int plane = 0; plane < 2; ++plane) {
        _Float16* p = qkv + rowoff + plane * H_;
        float x0 = (float)p[0], x1 = (float)p[1];
        p[0] = (_Float16)(x0 * c0 - x1 * s0);
        p[1] = (_Float16)(x1 * c1 + x0 * s1);
    }
}

// ---------------------------------------------------------------------------
// Sliding-window (W=512) causal attention, flash-style, WMMA.
// One block = 8 waves; each wave owns a 16-query tile of one (b,h).
// Q/K from qkv f16 [B*S,3072]; V from vT [B,NH,HD,S]; out -> attn f16.
// ---------------------------------------------------------------------------
__global__ __launch_bounds__(256) void attn_kernel(const _Float16* __restrict__ qkv,
                                                   const _Float16* __restrict__ vT,
                                                   _Float16* __restrict__ attn) {
    const int bid = blockIdx.x;
    const int qt = bid & 31;           // 32 tiles of 128 queries
    const int h  = (bid >> 5) & 15;
    const int b  = bid >> 9;
    const int wave = threadIdx.x >> 5;
    const int lane = threadIdx.x & 31;
    const int col  = lane & 15;
    const int lh   = lane >> 4;
    const int rowg = lh ? 8 : 0;
    const int q0   = qt * 128 + wave * 16;

    const _Float16* base  = qkv + (size_t)b * S_ * QKV_N;
    const _Float16* vbase = vT + ((size_t)(b * NH_ + h) * HD_) * S_;

    // Q fragments (two K-halves of d), pre-scaled by 1/sqrt(64)
    v16h aq[2];
    {
        int row = q0 + col;
        int kb = lh ? 8 : 0;
        const _Float16* qp = base + (size_t)row * QKV_N + h * HD_;
#pragma unroll
        for (int half = 0; half < 2; ++half) {
            int d0 = half * 32;
            v8h lo = *(const v8h*)(qp + d0 + kb);
            v8h hi = *(const v8h*)(qp + d0 + kb + 16);
            aq[half] = cat16(lo, hi);
#pragma unroll
            for (int t = 0; t < 16; ++t) aq[half][t] *= (_Float16)0.125f;
        }
    }

    float m[8], l[8];
#pragma unroll
    for (int r = 0; r < 8; ++r) { m[r] = -1e30f; l[r] = 0.f; }
    v8f o[4] = {};

    __shared__ _Float16 plds[8][16][40];
    _Float16 (*P)[40] = plds[wave];

    int jlo = q0 - (W_ - 1);
    if (jlo < 0) jlo = 0;
    jlo &= ~31;
    const int jhi = q0 + 16;

    for (int j0 = jlo; j0 < jhi; j0 += 32) {
        // ---- scores: two 16x16 tiles over 32 keys, contraction over d=64 ----
        v8f cs[2] = {};
#pragma unroll
        for (int half = 0; half < 2; ++half) {
            int d0 = half * 32;
#pragma unroll
            for (int ct = 0; ct < 2; ++ct) {
                int j = j0 + ct * 16 + col;
                if (j > S_ - 1) j = S_ - 1;
                int kb = lh ? 8 : 0;
                const _Float16* kp = base + (size_t)j * QKV_N + H_ + h * HD_ + d0 + kb;
                v8h lo = *(const v8h*)kp;
                v8h hi = *(const v8h*)(kp + 16);
                v16h bk = cat16(lo, hi);
                cs[ct] = __builtin_amdgcn_wmma_f32_16x16x32_f16(
                    false, aq[half], false, bk, (short)0, cs[ct], false, false);
            }
        }

        // ---- mask + online softmax (per-row, cross-lane reduce over 16 cols) ----
        float p0[8], p1[8];
#pragma unroll
        for (int r = 0; r < 8; ++r) {
            int q = q0 + rowg + r;
            int ja = j0 + col, jb = j0 + 16 + col;
            p0[r] = ((ja <= q) && (ja > q - W_)) ? cs[0][r] : -1e30f;
            p1[r] = ((jb <= q) && (jb > q - W_)) ? cs[1][r] : -1e30f;
        }
        float alpha[8];
#pragma unroll
        for (int r = 0; r < 8; ++r) {
            float t = fmaxf(p0[r], p1[r]);
#pragma unroll
            for (int mk = 1; mk < 16; mk <<= 1) t = fmaxf(t, __shfl_xor(t, mk, 32));
            float mnew = fmaxf(m[r], t);
            alpha[r] = __expf(m[r] - mnew);
            p0[r] = __expf(p0[r] - mnew);
            p1[r] = __expf(p1[r] - mnew);
            float rs = p0[r] + p1[r];
#pragma unroll
            for (int mk = 1; mk < 16; mk <<= 1) rs += __shfl_xor(rs, mk, 32);
            l[r] = l[r] * alpha[r] + rs;
            m[r] = mnew;
        }
#pragma unroll
        for (int nt = 0; nt < 4; ++nt)
#pragma unroll
            for (int r = 0; r < 8; ++r) o[nt][r] *= alpha[r];

        // ---- re-layout P (C-frag f32 -> A-frag f16) via per-wave LDS ----
#pragma unroll
        for (int r = 0; r < 8; ++r) {
            P[rowg + r][col]      = (_Float16)p0[r];
            P[rowg + r][16 + col] = (_Float16)p1[r];
        }
        asm volatile("s_wait_dscnt 0" ::: "memory");

        v16h ap;
        {
            int row = lane & 15;
            int kb = lh ? 8 : 0;
            v8h lo = *(const v8h*)&P[row][kb];
            v8h hi = *(const v8h*)&P[row][kb + 16];
            ap = cat16(lo, hi);
        }

        // ---- PV: O[16x64] += P[16x32] @ V[32x64], V from d-major vT ----
#pragma unroll
        for (int nt = 0; nt < 4; ++nt) {
            int d = nt * 16 + col;
            int kb = lh ? 8 : 0;
            const _Float16* vp = vbase + (size_t)d * S_ + j0 + kb;
            v8h lo = *(const v8h*)vp;        // masked tail rows read in-ws slack
            v8h hi = *(const v8h*)(vp + 16);
            v16h bv = cat16(lo, hi);
            o[nt] = __builtin_amdgcn_wmma_f32_16x16x32_f16(
                false, ap, false, bv, (short)0, o[nt], false, false);
        }
    }

    // ---- epilogue: normalize by l, write [B*S, H] ----
#pragma unroll
    for (int nt = 0; nt < 4; ++nt)
#pragma unroll
        for (int r = 0; r < 8; ++r) {
            int q = q0 + rowg + r;
            attn[((size_t)(b * S_ + q)) * H_ + h * HD_ + nt * 16 + col] =
                (_Float16)(o[nt][r] / l[r]);
        }
}

// ---------------------------------------------------------------------------
// Host-side orchestration
// ---------------------------------------------------------------------------
extern "C" void kernel_launch(void* const* d_in, const int* in_sizes, int n_in,
                              void* d_out, int out_size, void* d_ws, size_t ws_size,
                              hipStream_t stream) {
    (void)in_sizes; (void)n_in; (void)out_size; (void)ws_size;
    const float* x      = (const float*)d_in[0];
    const float* sinb   = (const float*)d_in[1];
    const float* cosb   = (const float*)d_in[2];
    const float* norm_w = (const float*)d_in[3];
    const float* w_qkv  = (const float*)d_in[4];
    const float* b_qkv  = (const float*)d_in[5];
    const float* w_in   = (const float*)d_in[6];
    const float* b_in   = (const float*)d_in[7];
    const float* w_out  = (const float*)d_in[8];
    const float* b_out  = (const float*)d_in[9];
    float* out = (float*)d_out;

    // workspace carve-up (f16 buffers, 256B aligned); vT not last (tail slack)
    char* w = (char*)d_ws;
    size_t off = 0;
    auto carve = [&](size_t bytes) {
        char* p = w + off;
        off = (off + bytes + 255) & ~(size_t)255;
        return p;
    };
    _Float16* xn_h   = (_Float16*)carve((size_t)M_ * H_ * 2);        // 16 MB
    _Float16* wqkv_h = (_Float16*)carve((size_t)H_ * QKV_N * 2);     //  6 MB
    _Float16* win_h  = (_Float16*)carve((size_t)H_ * I_ * 2);        //  8 MB
    _Float16* wout_h = (_Float16*)carve((size_t)(H_ + I_) * H_ * 2); // 10 MB
    _Float16* qkv_h  = (_Float16*)carve((size_t)M_ * QKV_N * 2);     // 48 MB
    _Float16* attn_h = (_Float16*)carve((size_t)M_ * H_ * 2);        // 16 MB
    _Float16* vT_h   = (_Float16*)carve((size_t)B_ * NH_ * HD_ * S_ * 2); // 16 MB
    _Float16* ff_h   = (_Float16*)carve((size_t)M_ * I_ * 2);        // 64 MB

    // 1) RMSNorm -> xn f16
    rmsnorm_kernel<<<M_, 256, 0, stream>>>(x, norm_w, xn_h);

    // 2) weight converts fp32 -> f16
    cvt_f32_f16<<<2048, 256, 0, stream>>>(w_qkv, wqkv_h, (long)H_ * QKV_N);
    cvt_f32_f16<<<2048, 256, 0, stream>>>(w_in,  win_h,  (long)H_ * I_);
    cvt_f32_f16<<<2048, 256, 0, stream>>>(w_out, wout_h, (long)(H_ + I_) * H_);

    // 3) QKV GEMM: [8192,1024] x [1024,3072] + b_qkv -> qkv f16
    gemm_wmma<0, 0, true, false, true><<<dim3(QKV_N / 128, M_ / 128), 256, 0, stream>>>(
        xn_h, wqkv_h, b_qkv, nullptr, qkv_h, M_, QKV_N, H_);

    // 4) RoPE in-place on q,k planes
    rope_kernel<<<(B_ * S_ * NH_ * (HD_ / 2)) / 256, 256, 0, stream>>>(qkv_h, sinb, cosb);

    // 5) V transpose to d-major
    transpose_v<<<(B_ * NH_ * HD_ * S_) / 256, 256, 0, stream>>>(qkv_h, vT_h);

    // 6) sliding-window attention -> attn f16 [8192,1024]
    attn_kernel<<<B_ * NH_ * (S_ / 128), 256, 0, stream>>>(qkv_h, vT_h, attn_h);

    // 7) FFN GEMM + exact GELU: [8192,1024] x [1024,4096] -> ff f16
    gemm_wmma<1, 0, true, false, true><<<dim3(I_ / 128, M_ / 128), 256, 0, stream>>>(
        xn_h, win_h, b_in, nullptr, ff_h, M_, I_, H_);

    // 8) out = attn @ w_out[:H] + b_out  (f32)
    gemm_wmma<0, 0, true, true, false><<<dim3(H_ / 128, M_ / 128), 256, 0, stream>>>(
        attn_h, wout_h, b_out, out, nullptr, M_, H_, H_);

    // 9) out += ff @ w_out[H:]
    gemm_wmma<0, 1, false, true, false><<<dim3(H_ / 128, M_ / 128), 256, 0, stream>>>(
        ff_h, wout_h + (size_t)H_ * H_, nullptr, out, nullptr, M_, H_, I_);
}
